// GraphTransformerMultiHeadAttentionLayer_43765716746403
// MI455X (gfx1250) — matrix operand order repaired and checked by reading, not compiled, not run on hardware
//
#include <hip/hip_runtime.h>
#include <hip/hip_bf16.h>

// ---------------------------------------------------------------------------
// GraphTransformer MHA for MI455X (gfx1250, wave32, WMMA).
// B=8 S=512 DM=1024 H=16 D=64. Output = concat(x[B,S,DM], attn[B,H,S,S]) f32.
// ---------------------------------------------------------------------------

#define Bsz   8
#define Ssz   512
#define DMsz  1024
#define Hsz   16
#define Dsz   64
#define XSIZE (Bsz * Ssz * DMsz)   // 4,194,304 floats

typedef __attribute__((ext_vector_type(16))) __bf16 v16bf;
typedef __attribute__((ext_vector_type(8)))  float  v8f;
typedef __attribute__((ext_vector_type(4)))  float  v4f;
typedef __attribute__((ext_vector_type(4)))  int    v4i;
typedef __attribute__((ext_vector_type(4)))  unsigned int v4u;
typedef __attribute__((ext_vector_type(2)))  unsigned int v2u;

union BF16x16 {
    unsigned short u[16];
    v4u            q[2];   // two 16-byte chunks
    v16bf          v;
};

static __device__ __forceinline__ unsigned short f2bf(float f) {
    unsigned int u = __float_as_uint(f);
    unsigned int r = u + 0x7FFFu + ((u >> 16) & 1u);   // round-to-nearest-even
    return (unsigned short)(r >> 16);
}

static __device__ __forceinline__ v8f wmma_bf16(const BF16x16& a, const BF16x16& b, v8f c) {
    return __builtin_amdgcn_wmma_f32_16x16x32_bf16(false, a.v, false, b.v,
                                                   (short)0, c, false, false);
}

// ---------------------------------------------------------------------------
// f32 -> bf16(u16) weight conversion, packed b64 stores
// ---------------------------------------------------------------------------
__global__ __launch_bounds__(256)
void cvt_kernel(const float* __restrict__ src, unsigned short* __restrict__ dst, int n) {
    int i = (blockIdx.x * 256 + threadIdx.x) * 4;
    if (i + 3 < n) {
        v4f v = *(const v4f*)(src + i);
        v2u p;
        p.x = (unsigned int)f2bf(v.x) | ((unsigned int)f2bf(v.y) << 16);
        p.y = (unsigned int)f2bf(v.z) | ((unsigned int)f2bf(v.w) << 16);
        *(v2u*)(dst + i) = p;
    }
}

// ---------------------------------------------------------------------------
// GEMM: out = A @ W^T + bias.   M=4096, N=1024, K=1024.
// W is bf16 row-major [N][K] => B-matrix column n == W row n (contiguous).
// Block 256 threads = 8 waves; tile 128x64; wave tile 16x64 (4 WMMA N-tiles).
// MODE 0: Q -> [B,H,S,D] bf16   MODE 1: K -> [B,H,S,D] bf16
// MODE 2: V -> [B,H,D,S] bf16 (transposed for PV B-fragments)
// MODE 3: A is bf16 ctx, out f32 [B,S,DM] (final projection)
// ---------------------------------------------------------------------------
template <int MODE>
__global__ __launch_bounds__(256)
void proj_kernel(const void* __restrict__ Aptr, const unsigned short* __restrict__ Wbf,
                 const float* __restrict__ bias, void* __restrict__ outp) {
    const int wave  = threadIdx.x >> 5;
    const int lane  = threadIdx.x & 31;
    const int m16   = lane & 15;
    const int hi    = lane >> 4;                 // 0 | 1
    const int rbase = blockIdx.x * 128 + wave * 16;
    const int nbase = blockIdx.y * 64;
    const int row   = rbase + m16;               // A-matrix row for this lane

    v8f acc[4] = {v8f{}, v8f{}, v8f{}, v8f{}};

    for (int ks = 0; ks < 32; ++ks) {            // K loop, 32 elems per step
        BF16x16 a;
        if (MODE < 3) {
            // f32 activations, convert to bf16 on load.
            // A lane layout: elems 0..7 -> K = ks*32 + hi*8 + j ; elems 8..15 -> +16
            const float* ap = (const float*)Aptr + (long)row * DMsz + ks * 32 + hi * 8;
            v4f x0 = *(const v4f*)(ap);
            v4f x1 = *(const v4f*)(ap + 4);
            v4f x2 = *(const v4f*)(ap + 16);
            v4f x3 = *(const v4f*)(ap + 20);
            #pragma unroll
            for (int j = 0; j < 4; ++j) {
                a.u[j]      = f2bf(x0[j]);
                a.u[4 + j]  = f2bf(x1[j]);
                a.u[8 + j]  = f2bf(x2[j]);
                a.u[12 + j] = f2bf(x3[j]);
            }
        } else {
            const unsigned short* ap =
                (const unsigned short*)Aptr + (long)row * DMsz + ks * 32 + hi * 8;
            a.q[0] = *(const v4u*)(ap);
            a.q[1] = *(const v4u*)(ap + 16);
        }
        #pragma unroll
        for (int nt = 0; nt < 4; ++nt) {
            // B lane layout: column n = lane%16, elems 0..7 -> K = ks*32 + hi*8 + j ; elems 8..15 -> +16
            const unsigned short* wp =
                Wbf + (long)(nbase + nt * 16 + m16) * DMsz + ks * 32 + hi * 8;
            BF16x16 bm;
            bm.q[0] = *(const v4u*)(wp);
            bm.q[1] = *(const v4u*)(wp + 16);
            acc[nt] = wmma_bf16(a, bm, acc[nt]);
        }
    }

    // C layout: lane col n = lane%16 ; VGPR i row = i + hi*8
    #pragma unroll
    for (int nt = 0; nt < 4; ++nt) {
        const int c  = nbase + nt * 16 + m16;
        const float bv = bias[c];
        #pragma unroll
        for (int i = 0; i < 8; ++i) {
            const int r   = rbase + i + hi * 8;
            const float v = acc[nt][i] + bv;
            if (MODE == 3) {
                ((float*)outp)[(long)r * DMsz + c] = v;
            } else {
                const int b = r >> 9, s = r & 511;
                const int h = c >> 6, d = c & 63;
                unsigned short* o = (unsigned short*)outp;
                if (MODE == 2)  // V transposed: [b,h,d,s]
                    o[(((long)(b * Hsz + h) * Dsz + d) << 9) | s] = f2bf(v);
                else            // Q/K: [b,h,s,d]
                    o[(((long)(b * Hsz + h) * Ssz + s) << 6) | d] = f2bf(v);
            }
        }
    }
}

// ---------------------------------------------------------------------------
// Fused attention. One block per (b, h, 16-row q tile); 128 threads = 4 waves.
//  Phase 1: energy = QK^T via WMMA -> LDS (raw accumulator).
//  Phase 2: row-parallel feature fusion (coalesced int4/float4 adjacency loads),
//           relu/combine/mask, softmax, f32 attention out (b128 stores),
//           bf16 probabilities into LDS.
//  Phase 3: ctx = P @ V via WMMA (A from LDS, B from V^T contiguous columns).
// ---------------------------------------------------------------------------
__global__ __launch_bounds__(128)
void attn_kernel(const unsigned short* __restrict__ Qbf,
                 const unsigned short* __restrict__ Kbf,
                 const unsigned short* __restrict__ Vt,
                 const int*   __restrict__ dadj,  const float* __restrict__ sadj,
                 const int*   __restrict__ dcadj, const int*   __restrict__ coadj,
                 const int*   __restrict__ mask,
                 const float* __restrict__ dist_e, const float* __restrict__ disc_e,
                 const float* __restrict__ cooc_e,
                 const float* __restrict__ fcw, const float* __restrict__ fcb,
                 const float* __restrict__ cw,  const float* __restrict__ cb,
                 float* __restrict__ attn_out, unsigned short* __restrict__ ctx) {
    __shared__ float          sc[16][520];      // scores / probs (f32)
    __shared__ unsigned short pb[16][520];      // probs bf16 for PV WMMA
    __shared__ float t_dist[1024];
    __shared__ float t_disc[17];
    __shared__ float t_cooc[6];

    const int tid  = threadIdx.x;
    const int wave = tid >> 5, lane = tid & 31;
    const int m16  = lane & 15, hi = lane >> 4;
    const int qt = blockIdx.x, h = blockIdx.y, b = blockIdx.z;
    const int bh = b * Hsz + h;

    for (int i = tid; i < 1023; i += 128) t_dist[i] = dist_e[i];
    if (tid < 17) t_disc[tid] = disc_e[tid];
    if (tid < 6)  t_cooc[tid] = cooc_e[tid];
    __syncthreads();

    const float NEG_INF = -__builtin_inff();

    // ---- Phase 1: raw energy tiles -> LDS ----
    {
        // Q fragments (reused across all 8 N-tiles); 2 k-steps over D=64
        const unsigned short* qrow = Qbf + (((long)bh * Ssz + qt * 16 + m16) << 6);
        BF16x16 a0, a1;
        a0.q[0] = *(const v4u*)(qrow + hi * 8);
        a0.q[1] = *(const v4u*)(qrow + hi * 8 + 16);
        a1.q[0] = *(const v4u*)(qrow + 32 + hi * 8);
        a1.q[1] = *(const v4u*)(qrow + 32 + hi * 8 + 16);

        const unsigned short* kbase = Kbf + (((long)bh * Ssz) << 6);

        #pragma unroll 1
        for (int it = 0; it < 8; ++it) {
            const int nt = wave + it * 4;
            const unsigned short* kp = kbase + ((long)(nt * 16 + m16) << 6);
            BF16x16 b0, b1;
            b0.q[0] = *(const v4u*)(kp + hi * 8);
            b0.q[1] = *(const v4u*)(kp + hi * 8 + 16);
            b1.q[0] = *(const v4u*)(kp + 32 + hi * 8);
            b1.q[1] = *(const v4u*)(kp + 32 + hi * 8 + 16);
            v8f acc = {};
            acc = wmma_bf16(a0, b0, acc);
            acc = wmma_bf16(a1, b1, acc);

            const int n = nt * 16 + m16;        // key index (C column of this lane)
            #pragma unroll
            for (int i = 0; i < 8; ++i) sc[i + hi * 8][n] = acc[i];
        }
    }
    __syncthreads();

    // ---- Phase 2: feature fusion + softmax (8 threads/row, 64 cols each) ----
    {
        const float fw0 = fcw[h * 4 + 0], fw1 = fcw[h * 4 + 1];
        const float fw2 = fcw[h * 4 + 2], fw3 = fcw[h * 4 + 3];
        const float fb = fcb[h], c0 = cw[0], c1 = cw[1], cbv = cb[0];

        const int row = tid >> 3, s8 = tid & 7;
        const int cb0 = s8 * 64;
        const int q   = qt * 16 + row;
        const long rb = ((long)(b * Ssz + q)) * Ssz + cb0;
        const int*   dp  = dadj  + rb;
        const float* sp  = sadj  + rb;
        const int*   dcp = dcadj + rb;
        const int*   cop = coadj + rb;
        const int*   mp  = mask + b * Ssz + cb0;

        float mx = NEG_INF;
        for (int j = 0; j < 64; j += 4) {
            const v4i d4  = *(const v4i*)(dp + j);
            const v4f s4  = *(const v4f*)(sp + j);
            const v4i dc4 = *(const v4i*)(dcp + j);
            const v4i co4 = *(const v4i*)(cop + j);
            const v4i m4  = *(const v4i*)(mp + j);
            #pragma unroll
            for (int u = 0; u < 4; ++u) {
                // unconditional LDS loads + cndmask selects (no exec branching)
                float feat = fw0 * t_dist[d4[u]] + fw1 * s4[u]
                           + fw2 * t_disc[dc4[u]] * (dc4[u] != 0 ? 1.f : 0.f)
                           + fw3 * t_cooc[co4[u]] * (co4[u] != 0 ? 1.f : 0.f) + fb;
                feat = fmaxf(feat, 0.f);
                float e = fmaxf(c0 * (sc[row][cb0 + j + u] * 0.125f) + c1 * feat + cbv, 0.f);
                e = (m4[u] == 0) ? NEG_INF : e;
                sc[row][cb0 + j + u] = e;
                mx = fmaxf(mx, e);
            }
        }
        mx = fmaxf(mx, __shfl_xor(mx, 1, 8));
        mx = fmaxf(mx, __shfl_xor(mx, 2, 8));
        mx = fmaxf(mx, __shfl_xor(mx, 4, 8));

        float sum = 0.f;
        for (int j = 0; j < 64; ++j) {
            float ex = (mx == NEG_INF) ? 0.f : __expf(sc[row][cb0 + j] - mx);
            sc[row][cb0 + j] = ex;
            sum += ex;
        }
        sum += __shfl_xor(sum, 1, 8);
        sum += __shfl_xor(sum, 2, 8);
        sum += __shfl_xor(sum, 4, 8);
        const float inv = sum > 0.f ? 1.f / sum : 0.f;

        float* arow = attn_out + (((long)bh * Ssz + q)) * Ssz + cb0;
        for (int j = 0; j < 64; j += 4) {
            v4f p;
            #pragma unroll
            for (int u = 0; u < 4; ++u) p[u] = sc[row][cb0 + j + u] * inv;
            *(v4f*)(arow + j) = p;                       // b128 store of attention
            v2u pk;
            pk.x = (unsigned int)f2bf(p[0]) | ((unsigned int)f2bf(p[1]) << 16);
            pk.y = (unsigned int)f2bf(p[2]) | ((unsigned int)f2bf(p[3]) << 16);
            *(v2u*)(&pb[row][cb0 + j]) = pk;             // b64 LDS store of probs
        }
    }
    __syncthreads();

    // ---- Phase 3: ctx = P @ V : each wave owns a 16-wide d tile; K=512 ----
    {
        v8f acc = {};
        const int dcol = wave * 16 + m16;       // output head-dim column
        const unsigned short* vp = Vt + (((long)bh * Dsz + dcol) << 9);
        #pragma unroll 1
        for (int ks = 0; ks < 16; ++ks) {
            BF16x16 a, bm;
            const unsigned short* pp = &pb[m16][ks * 32 + hi * 8];
            a.q[0] = *(const v4u*)(pp);
            a.q[1] = *(const v4u*)(pp + 16);
            const unsigned short* vv = vp + ks * 32 + hi * 8;
            bm.q[0] = *(const v4u*)(vv);
            bm.q[1] = *(const v4u*)(vv + 16);
            acc = wmma_bf16(a, bm, acc);
        }
        #pragma unroll
        for (int i = 0; i < 8; ++i) {
            const int mrow = i + hi * 8;
            ctx[((long)(b * Ssz + qt * 16 + mrow)) * DMsz + h * Dsz + dcol] = f2bf(acc[i]);
        }
    }
}

// ---------------------------------------------------------------------------
extern "C" void kernel_launch(void* const* d_in, const int* in_sizes, int n_in,
                              void* d_out, int out_size, void* d_ws, size_t ws_size,
                              hipStream_t stream) {
    const float* query = (const float*)d_in[0];
    const float* key   = (const float*)d_in[1];
    const float* value = (const float*)d_in[2];
    const int*   dadj  = (const int*)d_in[3];
    const float* sadj  = (const float*)d_in[4];
    const int*   dcadj = (const int*)d_in[5];
    const int*   coadj = (const int*)d_in[6];
    const int*   mask  = (const int*)d_in[7];
    const float* Wq = (const float*)d_in[8];  const float* bq = (const float*)d_in[9];
    const float* Wk = (const float*)d_in[10]; const float* bk = (const float*)d_in[11];
    const float* Wv = (const float*)d_in[12]; const float* bv = (const float*)d_in[13];
    const float* Wo = (const float*)d_in[14]; const float* bo = (const float*)d_in[15];
    const float* dist_e = (const float*)d_in[16];
    const float* disc_e = (const float*)d_in[17];
    const float* cooc_e = (const float*)d_in[18];
    const float* fcw = (const float*)d_in[19];
    const float* fcb = (const float*)d_in[20];
    const float* cw  = (const float*)d_in[21];
    const float* cb  = (const float*)d_in[22];

    // Workspace layout (bf16 stored as u16):
    char* ws = (char*)d_ws;
    const size_t WSZ = (size_t)DMsz * DMsz * 2;               // 2 MB per weight
    const size_t QSZ = (size_t)Bsz * Ssz * DMsz * 2;          // 8 MB per tensor
    unsigned short* wq_bf = (unsigned short*)(ws + 0 * WSZ);
    unsigned short* wk_bf = (unsigned short*)(ws + 1 * WSZ);
    unsigned short* wv_bf = (unsigned short*)(ws + 2 * WSZ);
    unsigned short* wo_bf = (unsigned short*)(ws + 3 * WSZ);
    unsigned short* Qbf   = (unsigned short*)(ws + 4 * WSZ);
    unsigned short* Kbf   = (unsigned short*)(ws + 4 * WSZ + 1 * QSZ);
    unsigned short* Vt    = (unsigned short*)(ws + 4 * WSZ + 2 * QSZ);
    unsigned short* ctx   = (unsigned short*)(ws + 4 * WSZ + 3 * QSZ);

    const int nW = DMsz * DMsz;
    cvt_kernel<<<nW / 1024, 256, 0, stream>>>(Wq, wq_bf, nW);
    cvt_kernel<<<nW / 1024, 256, 0, stream>>>(Wk, wk_bf, nW);
    cvt_kernel<<<nW / 1024, 256, 0, stream>>>(Wv, wv_bf, nW);
    cvt_kernel<<<nW / 1024, 256, 0, stream>>>(Wo, wo_bf, nW);

    dim3 pgrid(Bsz * Ssz / 128, DMsz / 64);                   // 32 x 16
    proj_kernel<0><<<pgrid, 256, 0, stream>>>(query, wq_bf, bq, Qbf);
    proj_kernel<1><<<pgrid, 256, 0, stream>>>(key,   wk_bf, bk, Kbf);
    proj_kernel<2><<<pgrid, 256, 0, stream>>>(value, wv_bf, bv, Vt);

    dim3 agrid(Ssz / 16, Hsz, Bsz);                           // 32 x 16 x 8
    attn_kernel<<<agrid, 128, 0, stream>>>(Qbf, Kbf, Vt, dadj, sadj, dcadj, coadj,
                                           mask, dist_e, disc_e, cooc_e,
                                           fcw, fcb, cw, cb,
                                           (float*)d_out + XSIZE, ctx);

    proj_kernel<3><<<pgrid, 256, 0, stream>>>(ctx, wo_bf, bo, d_out);
}
